// BlockRC2_16140487099014
// MI455X (gfx1250) — compile-verified, matrix operand order checked by reading, not run on hardware
//
#include <hip/hip_runtime.h>
#include <hip/hip_bf16.h>
#include <math.h>

// ---------------------------------------------------------------------------
// MI455X (gfx1250, wave32) implementation of the PRM + Token-Performer block.
// All dense GEMMs (implicit-GEMM dilated convs, kqv, MLP) use
// V_WMMA_F32_16X16X32_F16 with f32 accumulation; skinny performer pieces are
// VALU with LDS accumulation.  Fused epilogues keep each intermediate to one
// HBM round-trip (pipeline is ~bandwidth-bound at 23.3 TB/s).
// Conv reduction axis ordered k = kid*64 + ic so each 32-wide WMMA K-step has
// a constant kernel tap -> division-free im2col addressing (SALU-hoisted).
// ---------------------------------------------------------------------------

typedef _Float16 half_t;
typedef __attribute__((ext_vector_type(16))) _Float16 v16h;
typedef __attribute__((ext_vector_type(8)))  float    v8f;

// Problem constants (match reference)
#define BATCH   32
#define CIN     64
#define IMGSZ   128
#define NPIX    16384     // 128*128
#define TOKN    4096      // 64*64 post-stride-2 tokens
#define EBR     64
#define DIMC    192       // 3*EBR
#define EMB     64
#define MF      32        // performer features
#define INVSQRTM 0.17677669529663687f

// Workspace layout (bytes). Buffers reused once their producers/consumers
// are done:  XH region -> KP/QP,  TOK region -> YATT.
#define XH_OFF    ((size_t)0)                                   // f16 [B][CIN][128][128]  67.1MB
#define KP_OFF    ((size_t)0)                                   // f32 [B][T][32]          (reuse XH)
#define QP_OFF    ((size_t)16777216)                            // f32 [B][T][32]
#define TOK_OFF   ((size_t)67108864)                            // f16 [B][T][192]         50.3MB
#define YATT_OFF  TOK_OFF                                       // f32 [B][T][64]          (reuse TOK)
#define HLN_OFF   ((size_t)(67108864 + 50331648))               // f16 [B][T][192]
#define KQV_OFF   ((size_t)(67108864 + 2*50331648))             // f32 [B][T][192]         100.7MB
#define KSUM_OFF  (KQV_OFF + (size_t)BATCH*TOKN*DIMC*4)         // f32 [B][32]
#define KPTV_OFF  (KSUM_OFF + (size_t)4096)                     // f32 [B][64][32]

__device__ __forceinline__ float gelu_exact(float x) {
  return 0.5f * x * (1.0f + erff(x * 0.70710678118654752f));
}

__device__ __forceinline__ float wave_sum32(float v) {
  #pragma unroll
  for (int off = 16; off > 0; off >>= 1) v += __shfl_xor(v, off, 32);
  return v;
}

// Load a 16x32 f16 WMMA fragment (A row-major, or B assembled as [N][K]).
// tile: LDS matrix with K contiguous per row; stride in half elements.
// ISA layout: lanes 0-15 hold K0-7 (v0-3) and K16-23 (v4-7); lanes 16-31
// hold K8-15 and K24-31; M/N = lane&15 (+row0).
__device__ __forceinline__ v16h load_frag(const half_t* tile, int row0,
                                          int stride, int lane) {
  const half_t* p = tile + (row0 + (lane & 15)) * stride;
  const int hi = lane >> 4;
  v16h f;
  #pragma unroll
  for (int j = 0; j < 8; ++j) {
    const int kb = ((j >> 2) << 4) + (hi << 3) + ((j & 3) << 1);
    f[2 * j]     = p[kb];
    f[2 * j + 1] = p[kb + 1];
  }
  return f;
}

// ---------------------------------------------------------------------------
// K1: x [B,N,C] f32  ->  xh [B,C,H,W] f16  (64x64 LDS-tiled transpose)
// ---------------------------------------------------------------------------
__global__ __launch_bounds__(256) void k_pack(const float* __restrict__ x,
                                              half_t* __restrict__ xh) {
  __shared__ half_t tile[64][65];
  const int b = blockIdx.x >> 8;          // 256 n-tiles per batch
  const int nbase = (blockIdx.x & 255) << 6;
  const int tid = threadIdx.x;
  #pragma unroll
  for (int rep = 0; rep < 16; ++rep) {
    const int e = tid + (rep << 8);
    const int nn = e >> 6, cc = e & 63;
    tile[nn][cc] = (half_t)x[((size_t)b * NPIX + nbase + nn) * CIN + cc];
  }
  __syncthreads();
  #pragma unroll
  for (int rep = 0; rep < 16; ++rep) {
    const int e = tid + (rep << 8);
    const int cc = e >> 6, nn = e & 63;
    xh[((size_t)b * CIN + cc) * NPIX + nbase + nn] = tile[nn][cc];
  }
}

// ---------------------------------------------------------------------------
// K2: implicit-GEMM dilated conv + bias + GELU -> tokens f16 [B,T,192].
// Per block: 64 tokens x 64 out-channels, K=576 in 18 steps of 32 via WMMA.
// K axis ordered k = kid*64 + ic: each step has constant kernel tap (kid),
// so the im2col gather needs no integer division and A addresses are just
// base + kq*NPIX.  grid = (64 token tiles, 3 branches, 32 batches), 4 waves.
// ---------------------------------------------------------------------------
__global__ __launch_bounds__(128) void k_conv(const half_t* __restrict__ xh,
                                              const float* __restrict__ cw,
                                              const float* __restrict__ cb,
                                              half_t* __restrict__ tok) {
  __shared__ half_t As[64][34];
  __shared__ half_t Bs[64][34];
  const int tbase = blockIdx.x << 6;
  const int br = blockIdx.y;
  const int b  = blockIdx.z;
  const int dil = br + 1, pad = br + 1;
  const int tid = threadIdx.x, lane = tid & 31, wv = tid >> 5;
  const half_t* xb = xh + (size_t)b * CIN * NPIX;
  const float*  wb = cw + (size_t)br * EBR * CIN * 9;

  v8f acc[4];
  const v8f vzero = {};
  #pragma unroll
  for (int i = 0; i < 4; ++i) acc[i] = vzero;

  // Per-thread im2col coordinates (invariant over the K loop).
  const int a_tl = tid >> 5;              // token row handled in rep 0 (+4/rep)
  const int a_kq = tid & 31;              // hoisted: e>>5 grows by 4 per rep

  for (int kk = 0; kk < 18; ++kk) {
    const int ic0 = (kk & 1) << 5;        // 0 or 32
    const int kid = kk >> 1;              // kernel tap 0..8 (constant per step)
    const int kh = kid / 3, kw = kid - kh * 3;   // scalar (SALU)
    const int dh = kh * dil - pad, dw = kw * dil - pad;
    const half_t* xc = xb + ((size_t)ic0 << 14); // channel-block base

    // A tile: 64 tokens x 32 ic  (address = token base + kq*NPIX)
    #pragma unroll
    for (int rep = 0; rep < 16; ++rep) {
      const int tl = a_tl + (rep << 2);
      const int t = tbase + tl;
      const int oh = t >> 6, ow = t & 63;
      const int ih = (oh << 1) + dh;
      const int iw = (ow << 1) + dw;
      half_t v = (half_t)0.0f;
      if ((unsigned)ih < 128u && (unsigned)iw < 128u)
        v = xc[((size_t)a_kq << 14) + (ih << 7) + iw];
      As[tl][a_kq] = v;
    }
    // B tile as [oc][kq]: wb[(oc*64 + ic0+kq)*9 + kid]  (division-free)
    #pragma unroll
    for (int rep = 0; rep < 16; ++rep) {
      const int e = tid + (rep << 7);
      const int oc = e >> 5, kq = e & 31;
      Bs[oc][kq] = (half_t)wb[((oc << 6) + ic0 + kq) * 9 + kid];
    }
    if (kk < 17) {  // speculative prefetch of next K-step's channel block
      const int nic0 = ((kk + 1) & 1) << 5;
      __builtin_prefetch(xb + ((size_t)nic0 << 14) + ((size_t)a_kq << 14), 0, 3);
    }
    __syncthreads();
    const v16h va = load_frag(&As[0][0], wv << 4, 34, lane);
    #pragma unroll
    for (int nt = 0; nt < 4; ++nt) {
      const v16h vb = load_frag(&Bs[nt << 4][0], 0, 34, lane);
      acc[nt] = __builtin_amdgcn_wmma_f32_16x16x32_f16(
          false, va, false, vb, (short)0, acc[nt], false, false);
    }
    __syncthreads();
  }
  // Epilogue: bias + exact GELU, write tokens (branch-major channel concat)
  const int hi = lane >> 4, cn = lane & 15;
  #pragma unroll
  for (int nt = 0; nt < 4; ++nt) {
    const int oc = (nt << 4) + cn;
    const float bias = cb[br * EBR + oc];
    #pragma unroll
    for (int r = 0; r < 8; ++r) {
      const int row = (wv << 4) + r + (hi << 3);
      const int t = tbase + row;
      const float v = acc[nt][r] + bias;
      tok[((size_t)b * TOKN + t) * DIMC + br * EBR + oc] = (half_t)gelu_exact(v);
    }
  }
}

// ---------------------------------------------------------------------------
// K3: LayerNorm(192) over tokens -> hln f16.  One wave per token.
// ---------------------------------------------------------------------------
__global__ __launch_bounds__(128) void k_ln1(const half_t* __restrict__ tok,
                                             const float* __restrict__ g,
                                             const float* __restrict__ bt,
                                             half_t* __restrict__ hln) {
  const int token = (blockIdx.x << 2) + (threadIdx.x >> 5);
  const int lane = threadIdx.x & 31;
  const size_t base = (size_t)token * DIMC;
  float xv[6], s = 0.f, s2 = 0.f;
  #pragma unroll
  for (int j = 0; j < 6; ++j) {
    const float v = (float)tok[base + lane + (j << 5)];
    xv[j] = v; s += v; s2 += v * v;
  }
  s = wave_sum32(s); s2 = wave_sum32(s2);
  const float mean = s * (1.0f / DIMC);
  const float var  = s2 * (1.0f / DIMC) - mean * mean;
  const float rs = rsqrtf(var + 1e-5f);
  #pragma unroll
  for (int j = 0; j < 6; ++j) {
    const int c = lane + (j << 5);
    hln[base + c] = (half_t)((xv[j] - mean) * rs * g[c] + bt[c]);
  }
}

// ---------------------------------------------------------------------------
// K4: kqv GEMM  [131072 x 192] @ [192 x 192] + bias -> kqv f32.
// Per block: 64 rows x 64 cols, K=192 in 6 WMMA steps. grid=(2048,3).
// ---------------------------------------------------------------------------
__global__ __launch_bounds__(128) void k_kqv(const half_t* __restrict__ hln,
                                             const float* __restrict__ w,
                                             const float* __restrict__ bias,
                                             float* __restrict__ kqv) {
  __shared__ half_t As[64][34];
  __shared__ half_t Bs[64][34];
  const int rbase = blockIdx.x << 6;
  const int ct = blockIdx.y;
  const int tid = threadIdx.x, lane = tid & 31, wv = tid >> 5;

  v8f acc[4];
  const v8f vzero = {};
  #pragma unroll
  for (int i = 0; i < 4; ++i) acc[i] = vzero;

  for (int kk = 0; kk < 6; ++kk) {
    const int ks = kk << 5;
    #pragma unroll
    for (int rep = 0; rep < 16; ++rep) {
      const int e = tid + (rep << 7);
      const int rr = e >> 5, kq = e & 31;
      As[rr][kq] = hln[(size_t)(rbase + rr) * DIMC + ks + kq];
    }
    #pragma unroll
    for (int rep = 0; rep < 16; ++rep) {
      const int e = tid + (rep << 7);
      const int n = e >> 5, kq = e & 31;
      Bs[n][kq] = (half_t)w[(size_t)(ks + kq) * DIMC + (ct << 6) + n];
    }
    __syncthreads();
    const v16h va = load_frag(&As[0][0], wv << 4, 34, lane);
    #pragma unroll
    for (int nt = 0; nt < 4; ++nt) {
      const v16h vb = load_frag(&Bs[nt << 4][0], 0, 34, lane);
      acc[nt] = __builtin_amdgcn_wmma_f32_16x16x32_f16(
          false, va, false, vb, (short)0, acc[nt], false, false);
    }
    __syncthreads();
  }
  const int hi = lane >> 4, cn = lane & 15;
  #pragma unroll
  for (int nt = 0; nt < 4; ++nt) {
    const int col = (ct << 6) + (nt << 4) + cn;
    const float bv = bias[col];
    #pragma unroll
    for (int r = 0; r < 8; ++r) {
      const int row = rbase + (wv << 4) + r + (hi << 3);
      kqv[(size_t)row * DIMC + col] = acc[nt][r] + bv;
    }
  }
}

// ---------------------------------------------------------------------------
// K5: performer features  kp/qp = exp(w.t - |t|^2/2)/sqrt(m)  + ksum reduce.
// One wave per token (lane = feature m), 8 tokens per block.
// ---------------------------------------------------------------------------
__global__ __launch_bounds__(256) void k_perf(const float* __restrict__ kqv,
                                              const float* __restrict__ wperf,
                                              float* __restrict__ kp,
                                              float* __restrict__ qp,
                                              float* __restrict__ ksum) {
  __shared__ float wp[32][64];
  __shared__ float kb[8][64];
  __shared__ float qb[8][64];
  __shared__ float kpart[32];
  const int b  = blockIdx.x >> 9;
  const int tb = (blockIdx.x & 511) << 3;
  const int tid = threadIdx.x, wv = tid >> 5, lane = tid & 31;
  #pragma unroll
  for (int rep = 0; rep < 8; ++rep) {
    const int e = tid + (rep << 8);
    wp[e >> 6][e & 63] = wperf[e];
  }
  if (tid < 32) kpart[tid] = 0.f;
  const int t = tb + wv;
  const size_t ro = ((size_t)b * TOKN + t) * DIMC;
  kb[wv][lane]      = kqv[ro + lane];
  kb[wv][lane + 32] = kqv[ro + lane + 32];
  qb[wv][lane]      = kqv[ro + 64 + lane];
  qb[wv][lane + 32] = kqv[ro + 64 + lane + 32];
  __syncthreads();
  float dk = 0.f, dq = 0.f, xk = 0.f, xq = 0.f;
  #pragma unroll 8
  for (int d = 0; d < 64; ++d) {
    const float kv = kb[wv][d], qv = qb[wv][d], wr = wp[lane][d];
    dk += wr * kv; dq += wr * qv; xk += kv * kv; xq += qv * qv;
  }
  const float kpv = expf(dk - 0.5f * xk) * INVSQRTM;
  const float qpv = expf(dq - 0.5f * xq) * INVSQRTM;
  const size_t po = ((size_t)b * TOKN + t) * MF + lane;
  kp[po] = kpv; qp[po] = qpv;
  atomicAdd(&kpart[lane], kpv);
  __syncthreads();
  if (tid < 32) atomicAdd(&ksum[b * MF + tid], kpart[tid]);
}

// ---------------------------------------------------------------------------
// K6: kptv[b,d,m] = sum_t v[b,t,d]*kp[b,t,m].  Per-thread 8-cell register
// accumulation over 256 tokens, LDS-staged tiles, one global atomic per cell.
// ---------------------------------------------------------------------------
__global__ __launch_bounds__(256) void k_kptv(const float* __restrict__ kqv,
                                              const float* __restrict__ kp,
                                              float* __restrict__ kptv) {
  __shared__ float vbuf[32][65];
  __shared__ float kpb[32][33];
  const int b  = blockIdx.x >> 4;
  const int tb = (blockIdx.x & 15) << 8;
  const int tid = threadIdx.x;
  const int d = tid >> 2, m0 = (tid & 3) << 3;
  float acc[8];
  #pragma unroll
  for (int j = 0; j < 8; ++j) acc[j] = 0.f;
  for (int ch = 0; ch < 8; ++ch) {
    const int t0 = tb + (ch << 5);
    #pragma unroll
    for (int rep = 0; rep < 8; ++rep) {
      const int e = tid + (rep << 8);
      const int tt = e >> 6, dd = e & 63;
      vbuf[tt][dd] = kqv[((size_t)b * TOKN + t0 + tt) * DIMC + 128 + dd];
    }
    #pragma unroll
    for (int rep = 0; rep < 4; ++rep) {
      const int e = tid + (rep << 8);
      const int tt = e >> 5, mm = e & 31;
      kpb[tt][mm] = kp[((size_t)b * TOKN + t0 + tt) * MF + mm];
    }
    __syncthreads();
    #pragma unroll 4
    for (int tt = 0; tt < 32; ++tt) {
      const float vv = vbuf[tt][d];
      #pragma unroll
      for (int j = 0; j < 8; ++j) acc[j] += vv * kpb[tt][m0 + j];
    }
    __syncthreads();
  }
  #pragma unroll
  for (int j = 0; j < 8; ++j)
    atomicAdd(&kptv[((size_t)b * EMB + d) * MF + m0 + j], acc[j]);
}

// ---------------------------------------------------------------------------
// K7: yatt = v + proj(qp @ kptv^T / (qp.ksum + eps)).  One wave per token,
// kptv / proj_w / ksum cached in LDS.
// ---------------------------------------------------------------------------
__global__ __launch_bounds__(256) void k_mix(const float* __restrict__ kqv,
                                             const float* __restrict__ qp,
                                             const float* __restrict__ ksum,
                                             const float* __restrict__ kptv,
                                             const float* __restrict__ pw,
                                             const float* __restrict__ pb,
                                             float* __restrict__ yatt) {
  __shared__ float kl[64][32];
  __shared__ float pwl[64][64];
  __shared__ float pbl[64];
  __shared__ float ksl[32];
  __shared__ float ql[8][32];
  __shared__ float al[8][64];
  const int b  = blockIdx.x >> 9;
  const int tb = (blockIdx.x & 511) << 3;
  const int tid = threadIdx.x, wv = tid >> 5, lane = tid & 31;
  #pragma unroll
  for (int rep = 0; rep < 8; ++rep) {
    const int e = tid + (rep << 8);
    kl[e >> 5][e & 31] = kptv[(size_t)b * (EMB * MF) + e];
  }
  #pragma unroll
  for (int rep = 0; rep < 16; ++rep) {
    const int e = tid + (rep << 8);
    pwl[e >> 6][e & 63] = pw[e];
  }
  if (tid < 64) pbl[tid] = pb[tid];
  if (tid < 32) ksl[tid] = ksum[b * MF + tid];
  const int t = tb + wv;
  const float qv = qp[((size_t)b * TOKN + t) * MF + lane];
  ql[wv][lane] = qv;
  __syncthreads();
  const float D = wave_sum32(qv * ksl[lane]);
  #pragma unroll
  for (int h = 0; h < 2; ++h) {
    const int d = lane + (h << 5);
    float num = 0.f;
    #pragma unroll 8
    for (int m = 0; m < 32; ++m) num += ql[wv][m] * kl[d][m];
    al[wv][d] = num / (D + 1e-8f);
  }
  __syncthreads();
  const size_t row = (size_t)b * TOKN + t;
  #pragma unroll
  for (int h = 0; h < 2; ++h) {
    const int d = lane + (h << 5);
    float o = kqv[row * DIMC + 128 + d] + pbl[d];
    #pragma unroll 8
    for (int e = 0; e < 64; ++e) o += al[wv][e] * pwl[e][d];
    yatt[row * EMB + d] = o;
  }
}

// ---------------------------------------------------------------------------
// K8: out = yatt + W2 @ gelu(W1 @ LN(yatt) + b1) + b2.  Fully fused:
// LN2 in registers, both 64x64 GEMMs via WMMA with all operands in LDS.
// Per block: 64 rows, 128 threads (4 waves).
// ---------------------------------------------------------------------------
__global__ __launch_bounds__(128) void k_mlp(const float* __restrict__ yatt,
                                             const float* __restrict__ g,
                                             const float* __restrict__ bt,
                                             const float* __restrict__ w1,
                                             const float* __restrict__ b1,
                                             const float* __restrict__ w2,
                                             const float* __restrict__ b2,
                                             float* __restrict__ out) {
  __shared__ half_t Hs[64][66];
  __shared__ half_t W1s[64][66];
  __shared__ half_t W2s[64][66];
  __shared__ half_t T1s[64][66];
  __shared__ float b1l[64], b2l[64];
  const int rbase = blockIdx.x << 6;
  const int tid = threadIdx.x, lane = tid & 31, wv = tid >> 5;
  // LayerNorm(64): two threads per row, pair-reduce via shfl.
  {
    const int r = tid >> 1, hf = tid & 1;
    const float* p = yatt + ((size_t)(rbase + r) << 6) + (hf << 5);
    float xv[32], s = 0.f, s2 = 0.f;
    #pragma unroll
    for (int j = 0; j < 32; ++j) { const float v = p[j]; xv[j] = v; s += v; s2 += v * v; }
    s  += __shfl_xor(s, 1, 32);
    s2 += __shfl_xor(s2, 1, 32);
    const float mean = s * (1.0f / 64.f);
    const float var  = s2 * (1.0f / 64.f) - mean * mean;
    const float rs = rsqrtf(var + 1e-5f);
    #pragma unroll
    for (int j = 0; j < 32; ++j) {
      const int c = (hf << 5) + j;
      Hs[r][c] = (half_t)((xv[j] - mean) * rs * g[c] + bt[c]);
    }
  }
  #pragma unroll
  for (int rep = 0; rep < 32; ++rep) {
    const int e = tid + (rep << 7);
    const int n = e >> 6, k = e & 63;
    W1s[n][k] = (half_t)w1[(k << 6) + n];
    W2s[n][k] = (half_t)w2[(k << 6) + n];
  }
  if (tid < 64) { b1l[tid] = b1[tid]; b2l[tid] = b2[tid]; }
  __syncthreads();

  const v8f vzero = {};
  v8f acc[4];
  #pragma unroll
  for (int i = 0; i < 4; ++i) acc[i] = vzero;
  #pragma unroll
  for (int kk = 0; kk < 2; ++kk) {
    const int ks = kk << 5;
    const v16h va = load_frag(&Hs[0][ks], wv << 4, 66, lane);
    #pragma unroll
    for (int nt = 0; nt < 4; ++nt) {
      const v16h vb = load_frag(&W1s[nt << 4][ks], 0, 66, lane);
      acc[nt] = __builtin_amdgcn_wmma_f32_16x16x32_f16(
          false, va, false, vb, (short)0, acc[nt], false, false);
    }
  }
  const int hi = lane >> 4, cn = lane & 15;
  #pragma unroll
  for (int nt = 0; nt < 4; ++nt) {
    const int col = (nt << 4) + cn;
    #pragma unroll
    for (int r = 0; r < 8; ++r) {
      const int row = (wv << 4) + r + (hi << 3);
      T1s[row][col] = (half_t)gelu_exact(acc[nt][r] + b1l[col]);
    }
  }
  __syncthreads();
  v8f acc2[4];
  #pragma unroll
  for (int i = 0; i < 4; ++i) acc2[i] = vzero;
  #pragma unroll
  for (int kk = 0; kk < 2; ++kk) {
    const int ks = kk << 5;
    const v16h va = load_frag(&T1s[0][ks], wv << 4, 66, lane);
    #pragma unroll
    for (int nt = 0; nt < 4; ++nt) {
      const v16h vb = load_frag(&W2s[nt << 4][ks], 0, 66, lane);
      acc2[nt] = __builtin_amdgcn_wmma_f32_16x16x32_f16(
          false, va, false, vb, (short)0, acc2[nt], false, false);
    }
  }
  #pragma unroll
  for (int nt = 0; nt < 4; ++nt) {
    const int col = (nt << 4) + cn;
    const float bv = b2l[col];
    #pragma unroll
    for (int r = 0; r < 8; ++r) {
      const int row = (wv << 4) + r + (hi << 3);
      const size_t gi = ((size_t)(rbase + row) << 6) + col;
      out[gi] = yatt[gi] + acc2[nt][r] + bv;
    }
  }
}

// ---------------------------------------------------------------------------
extern "C" void kernel_launch(void* const* d_in, const int* in_sizes, int n_in,
                              void* d_out, int out_size, void* d_ws, size_t ws_size,
                              hipStream_t stream) {
  (void)in_sizes; (void)n_in; (void)out_size; (void)ws_size;
  const float* x      = (const float*)d_in[0];
  const float* conv_w = (const float*)d_in[1];
  const float* conv_b = (const float*)d_in[2];
  const float* ln1_g  = (const float*)d_in[3];
  const float* ln1_b  = (const float*)d_in[4];
  const float* kqv_w  = (const float*)d_in[5];
  const float* kqv_b  = (const float*)d_in[6];
  const float* proj_w = (const float*)d_in[7];
  const float* proj_b = (const float*)d_in[8];
  const float* w_perf = (const float*)d_in[9];
  const float* ln2_g  = (const float*)d_in[10];
  const float* ln2_b  = (const float*)d_in[11];
  const float* mlp_w1 = (const float*)d_in[12];
  const float* mlp_b1 = (const float*)d_in[13];
  const float* mlp_w2 = (const float*)d_in[14];
  const float* mlp_b2 = (const float*)d_in[15];

  char* ws = (char*)d_ws;
  half_t* xh   = (half_t*)(ws + XH_OFF);
  half_t* tok  = (half_t*)(ws + TOK_OFF);
  half_t* hln  = (half_t*)(ws + HLN_OFF);
  float*  kqv  = (float*)(ws + KQV_OFF);
  float*  kp   = (float*)(ws + KP_OFF);     // reuses XH region (xh dead)
  float*  qp   = (float*)(ws + QP_OFF);
  float*  ksum = (float*)(ws + KSUM_OFF);
  float*  kptv = (float*)(ws + KPTV_OFF);
  float*  yatt = (float*)(ws + YATT_OFF);   // reuses TOK region (tok dead)
  float*  out  = (float*)d_out;

  // Zero the atomic accumulators (ksum + kptv are contiguous).
  hipMemsetAsync(ws + KSUM_OFF, 0, 4096 + (size_t)BATCH * EMB * MF * 4, stream);

  k_pack<<<dim3(BATCH * 256), dim3(256), 0, stream>>>(x, xh);
  k_conv<<<dim3(64, 3, BATCH), dim3(128), 0, stream>>>(xh, conv_w, conv_b, tok);
  k_ln1 <<<dim3((BATCH * TOKN) / 4), dim3(128), 0, stream>>>(tok, ln1_g, ln1_b, hln);
  k_kqv <<<dim3((BATCH * TOKN) / 64, 3), dim3(128), 0, stream>>>(hln, kqv_w, kqv_b, kqv);
  k_perf<<<dim3(BATCH * 512), dim3(256), 0, stream>>>(kqv, w_perf, kp, qp, ksum);
  k_kptv<<<dim3(BATCH * 16), dim3(256), 0, stream>>>(kqv, kp, kptv);
  k_mix <<<dim3(BATCH * 512), dim3(256), 0, stream>>>(kqv, qp, ksum, kptv,
                                                      proj_w, proj_b, yatt);
  k_mlp <<<dim3((BATCH * TOKN) / 64), dim3(128), 0, stream>>>(
      yatt, ln2_g, ln2_b, mlp_w1, mlp_b1, mlp_w2, mlp_b2, out);
}
